// KPPRNet_52312701665291
// MI455X (gfx1250) — matrix operand order repaired and checked by reading, not compiled, not run on hardware
//
#include <hip/hip_runtime.h>
#include <math.h>
#include <stdint.h>

// ---------------------------------------------------------------------------
// KPPRNet forward for MI455X (gfx1250, wave32, WMMA).
// - All dense contractions: v_wmma_f32_16x16x32_bf16 (f32 accumulate).
// - B operands pre-packed into wave32 B-fragment layout (32B contiguous/lane).
// - Kernel-point weights stored directly in A-fragment layout.
// - KPConv fused: async memory->LDS neighbor gather (global_load_async_to_lds,
//   ASYNCcnt, double-buffered) -> K=32 WMMA -> LDS agg -> 960-deep WMMA GEMM.
// - GEMM inner loops software-pipelined (load k+1 while WMMA'ing k).
// Everything is L2-resident (tens of MB vs 192MB L2); HBM is irrelevant here.
// NOTE: input m (bool) is read as 1-byte values.
// ---------------------------------------------------------------------------

typedef __attribute__((ext_vector_type(16))) __bf16 v16bf;
typedef __attribute__((ext_vector_type(8)))  __bf16 v8bf;
typedef __attribute__((ext_vector_type(8)))  float  v8f;

#define BB 8
#define NN 2048
#define BN 16384
#define KNBR 32
#define KPTS 15
#define SLOPE 0.1f

static __device__ __forceinline__ __bf16 f2b(float f) {
  unsigned u = __builtin_bit_cast(unsigned, f);
  unsigned r = (u + 0x7FFFu + ((u >> 16) & 1u)) >> 16;
  unsigned short s = (unsigned short)r;
  return __builtin_bit_cast(__bf16, s);
}

static __device__ __forceinline__ v16bf cat16(v8bf lo, v8bf hi) {
  return __builtin_shufflevector(lo, hi, 0,1,2,3,4,5,6,7,8,9,10,11,12,13,14,15);
}

static __device__ __forceinline__ v8f wmma_bf16(v16bf a, v16bf b, v8f c) {
  return __builtin_amdgcn_wmma_f32_16x16x32_bf16(false, a, false, b, (short)0, c,
                                                 false, false);
}

// A-fragment load from row-major bf16: lane<16 gets K {k0..k0+7, k0+16..k0+23},
// lane>=16 gets {k0+8..k0+15, k0+24..k0+31} (16-bit A interleave).
static __device__ __forceinline__ v16bf loadA(const __bf16* arow, int k0, int hi) {
  v8bf lo = *(const v8bf*)(arow + k0 + hi * 8);
  v8bf hs = *(const v8bf*)(arow + k0 + 16 + hi * 8);
  return cat16(lo, hs);
}

// Async copy of one 128B row (64 bf16) from global to LDS; tracked by ASYNCcnt.
// The 24-bit inst offset applies to both LDS and global addresses.
static __device__ __forceinline__ void async_row_gather(const __bf16* src,
                                                        void* ldsdst) {
  unsigned ldsa = (unsigned)(uintptr_t)ldsdst;   // low 32 bits = LDS offset
  unsigned long long ga = (unsigned long long)(uintptr_t)src;
  asm volatile(
      "global_load_async_to_lds_b128 %0, %1, off\n\t"
      "global_load_async_to_lds_b128 %0, %1, off offset:16\n\t"
      "global_load_async_to_lds_b128 %0, %1, off offset:32\n\t"
      "global_load_async_to_lds_b128 %0, %1, off offset:48\n\t"
      "global_load_async_to_lds_b128 %0, %1, off offset:64\n\t"
      "global_load_async_to_lds_b128 %0, %1, off offset:80\n\t"
      "global_load_async_to_lds_b128 %0, %1, off offset:96\n\t"
      "global_load_async_to_lds_b128 %0, %1, off offset:112"
      :: "v"(ldsa), "v"(ga) : "memory");
}

static __device__ __forceinline__ float wredsum(float v) {
  #pragma unroll
  for (int o = 16; o > 0; o >>= 1) v += __shfl_xor(v, o, 32);
  return v;
}
static __device__ __forceinline__ float wredmax(float v) {
  #pragma unroll
  for (int o = 16; o > 0; o >>= 1) v = fmaxf(v, __shfl_xor(v, o, 32));
  return v;
}

// ---------------------------------------------------------------------------
// Pack a row-major f32 [K,N] matrix into WMMA B-fragment bf16 layout:
//   out[((ks*nt+ct)*32+lane)*16+h] = W[ks*32+(h/8)*16+(lane/16)*8+h%8][ct*16+lane%16]
// ---------------------------------------------------------------------------
__global__ void pack_b_kernel(const float* __restrict__ W, __bf16* __restrict__ out,
                              int K, int N) {
  int t = blockIdx.x * blockDim.x + threadIdx.x;
  int ntiles = N >> 4;
  int total = (K >> 5) * ntiles * 512;
  if (t >= total) return;
  int h    = t & 15;
  int lane = (t >> 4) & 31;
  int ct   = (t >> 9) % ntiles;
  int ks   = (t >> 9) / ntiles;
  int k    = (ks << 5) + ((h >> 3) << 4) + ((lane >> 4) << 3) + (h & 7);
  int col  = (ct << 4) + (lane & 15);
  out[t] = f2b(W[(size_t)k * N + col]);
}

// ---------------------------------------------------------------------------
// Generic WMMA GEMM:  C = act(A1@B1 [+ A2@B2]), software-pipelined.
//   A*: row-major bf16 [M,K]; B*: packed fragment; one wave per 16x16 tile.
// ---------------------------------------------------------------------------
__global__ __launch_bounds__(128)
void gemm_bf16_kernel(const __bf16* __restrict__ A1, const __bf16* __restrict__ B1,
                      int K1,
                      const __bf16* __restrict__ A2, const __bf16* __restrict__ B2,
                      int K2, int M, int N, int act,
                      __bf16* __restrict__ Cbf, float* __restrict__ Cf) {
  const int lane = threadIdx.x & 31;
  const int wave = blockIdx.x * (blockDim.x >> 5) + (threadIdx.x >> 5);
  const int ntiles = N >> 4;
  const int tiles = (M >> 4) * ntiles;
  if (wave >= tiles) return;
  const int mt = wave / ntiles, ct = wave % ntiles;
  const int row = (mt << 4) + (lane & 15);
  const int hi = lane >> 4;
  const size_t bstr = (size_t)ntiles * 512;
  v8f acc = {};
  {
    const __bf16* arow = A1 + (size_t)row * K1;
    const __bf16* bb = B1 + ((size_t)ct * 32 + lane) * 16;
    const int ksteps = K1 >> 5;
    v16bf ac = loadA(arow, 0, hi);
    v16bf bc = *(const v16bf*)(bb);
    for (int ks = 0; ks < ksteps - 1; ++ks) {
      v16bf an = loadA(arow, (ks + 1) << 5, hi);
      v16bf bn = *(const v16bf*)(bb + (size_t)(ks + 1) * bstr);
      acc = wmma_bf16(ac, bc, acc);
      ac = an; bc = bn;
    }
    acc = wmma_bf16(ac, bc, acc);
  }
  if (A2 != nullptr) {
    const __bf16* arow = A2 + (size_t)row * K2;
    const __bf16* bb = B2 + ((size_t)ct * 32 + lane) * 16;
    const int ksteps = K2 >> 5;
    v16bf ac = loadA(arow, 0, hi);
    v16bf bc = *(const v16bf*)(bb);
    for (int ks = 0; ks < ksteps - 1; ++ks) {
      v16bf an = loadA(arow, (ks + 1) << 5, hi);
      v16bf bn = *(const v16bf*)(bb + (size_t)(ks + 1) * bstr);
      acc = wmma_bf16(ac, bc, acc);
      ac = an; bc = bn;
    }
    acc = wmma_bf16(ac, bc, acc);
  }
  const int col = (ct << 4) + (lane & 15);
  #pragma unroll
  for (int r = 0; r < 8; ++r) {
    float v = acc[r];
    if (act == 1) v = v > 0.f ? v : SLOPE * v;
    int rr = (mt << 4) + r + 8 * hi;
    if (Cbf) Cbf[(size_t)rr * N + col] = f2b(v);
    if (Cf)  Cf[(size_t)rr * N + col] = v;
  }
}

// ---------------------------------------------------------------------------
// Masked coords
// ---------------------------------------------------------------------------
__global__ void coords_kernel(const float* __restrict__ x,
                              const unsigned char* __restrict__ m,
                              float* __restrict__ coords) {
  int i = blockIdx.x * blockDim.x + threadIdx.x;
  if (i >= BN) return;
  bool mm = m[i] != 0;
  #pragma unroll
  for (int d = 0; d < 3; ++d)
    coords[(size_t)i * 3 + d] = mm ? 1.0e6f : x[(size_t)i * 3 + d];
}

// ---------------------------------------------------------------------------
// PointNet pointwise MLP 3->64->64 (relu), output bf16
// ---------------------------------------------------------------------------
__global__ __launch_bounds__(256)
void pointnet_kernel(const float* __restrict__ x,
                     const float* __restrict__ w1, const float* __restrict__ b1,
                     const float* __restrict__ w2, const float* __restrict__ b2,
                     __bf16* __restrict__ fbf) {
  __shared__ float sw1[192], sb1[64], sw2[4096], sb2[64];
  int t = threadIdx.x;
  for (int i = t; i < 192; i += 256) sw1[i] = w1[i];
  if (t < 64) { sb1[t] = b1[t]; sb2[t] = b2[t]; }
  for (int i = t; i < 4096; i += 256) sw2[i] = w2[i];
  __syncthreads();
  int g = blockIdx.x * 256 + t;
  float p0 = x[(size_t)g * 3], p1 = x[(size_t)g * 3 + 1], p2 = x[(size_t)g * 3 + 2];
  float h1[64];
  #pragma unroll
  for (int c = 0; c < 64; ++c) {
    float v = fmaf(p0, sw1[c], fmaf(p1, sw1[64 + c], fmaf(p2, sw1[128 + c], sb1[c])));
    h1[c] = v > 0.f ? v : 0.f;
  }
  for (int c = 0; c < 64; ++c) {
    float acc = sb2[c];
    #pragma unroll 8
    for (int j = 0; j < 64; ++j) acc = fmaf(h1[j], sw2[j * 64 + c], acc);
    fbf[(size_t)g * 64 + c] = f2b(acc > 0.f ? acc : 0.f);
  }
}

// ---------------------------------------------------------------------------
// KNN: one workgroup (256 thr) per query point, iterative argmin over LDS
// ---------------------------------------------------------------------------
__global__ __launch_bounds__(256)
void knn_kernel(const float* __restrict__ coords, int* __restrict__ idx) {
  __shared__ float d2[NN];
  __shared__ float rv[256];
  __shared__ int ri[256];
  int g = blockIdx.x;
  int b = g / NN;
  int t = threadIdx.x;
  const float* cb = coords + (size_t)b * NN * 3;
  float qx = coords[(size_t)g * 3], qy = coords[(size_t)g * 3 + 1],
        qz = coords[(size_t)g * 3 + 2];
  for (int j = t; j < NN; j += 256) {
    float dx = cb[(size_t)j * 3] - qx;
    float dy = cb[(size_t)j * 3 + 1] - qy;
    float dz = cb[(size_t)j * 3 + 2] - qz;
    d2[j] = dx * dx + dy * dy + dz * dz;
  }
  __syncthreads();
  for (int s = 0; s < KNBR; ++s) {
    float bv = 3.0e38f; int bi = 0;
    for (int j = t; j < NN; j += 256) {
      float v = d2[j];
      if (v < bv) { bv = v; bi = j; }
    }
    rv[t] = bv; ri[t] = bi;
    __syncthreads();
    for (int off = 128; off > 0; off >>= 1) {
      if (t < off && rv[t + off] < rv[t]) { rv[t] = rv[t + off]; ri[t] = ri[t + off]; }
      __syncthreads();
    }
    if (t == 0) {
      idx[(size_t)g * KNBR + s] = b * NN + ri[0];
      d2[ri[0]] = 3.0e38f;
    }
    __syncthreads();
  }
}

// ---------------------------------------------------------------------------
// Kernel-point weights, written directly in WMMA A-fragment layout.
// ---------------------------------------------------------------------------
__global__ __launch_bounds__(128)
void kpw_kernel(const float* __restrict__ coords, const int* __restrict__ idx,
                const float* __restrict__ kp, __bf16* __restrict__ kpw) {
  int lane = threadIdx.x & 31;
  int g = blockIdx.x * 4 + (threadIdx.x >> 5);
  int M = lane & 15;
  int hi = lane >> 4;
  float qx = coords[(size_t)g * 3], qy = coords[(size_t)g * 3 + 1],
        qz = coords[(size_t)g * 3 + 2];
  float kx = 0.f, ky = 0.f, kz = 0.f;
  if (M < KPTS) { kx = kp[M * 3]; ky = kp[M * 3 + 1]; kz = kp[M * 3 + 2]; }
  v16bf out;
  #pragma unroll
  for (int h = 0; h < 16; ++h) {
    int nbr = hi * 16 + h;
    int nidx = idx[(size_t)g * KNBR + nbr];
    float dx = coords[(size_t)nidx * 3] - qx;
    float dy = coords[(size_t)nidx * 3 + 1] - qy;
    float dz = coords[(size_t)nidx * 3 + 2] - qz;
    float w = 0.f;
    if (M < KPTS) {
      float ex = dx - kx, ey = dy - ky, ez = dz - kz;
      float dist = sqrtf(ex * ex + ey * ey + ez * ez);
      w = fmaxf(0.f, 1.f - dist * 2.0f);  // 1 - dist/0.5
    }
    out[h] = f2b(w);
  }
  *(v16bf*)(kpw + ((size_t)g * 32 + lane) * 16) = out;
}

// ---------------------------------------------------------------------------
// Fused KPConv: 16 points per workgroup (4 waves).
// Stage 1 (per point, per wave): async-gather 32 neighbor x1 rows straight
// into LDS (double-buffered, ASYNCcnt), agg = w@fn via K=32 WMMA per 16-col
// tile, stash agg (bf16) into LDS [16][960] (rows wave-disjoint).
// Stage 2: x2 = lrelu(Agg[16,960] @ WkPack[960,64]), 30 pipelined WMMA steps.
// ---------------------------------------------------------------------------
__global__ __launch_bounds__(128)
void kpconv_kernel(const __bf16* __restrict__ x1, const int* __restrict__ idx,
                   const __bf16* __restrict__ kpw, const __bf16* __restrict__ wkp,
                   __bf16* __restrict__ x2) {
  __shared__ __align__(16) __bf16 gbuf[4][2][32][72];  // per-wave double buffer
  __shared__ __align__(16) __bf16 agg[16][960];
  const int lane = threadIdx.x & 31;
  const int w = threadIdx.x >> 5;
  const int gbase = blockIdx.x * 16;
  const int hi = lane >> 4;
  const int cl = lane & 15;
  const int kb = hi * 16;

  // prologue: async gather for point 0 of this wave
  {
    int gg = gbase + w * 4;
    int n = idx[(size_t)gg * KNBR + lane];
    async_row_gather(x1 + (size_t)n * 64, &gbuf[w][0][lane][0]);
  }
  for (int pp = 0; pp < 4; ++pp) {
    if (pp < 3) {
      int gg = gbase + w * 4 + pp + 1;
      int n = idx[(size_t)gg * KNBR + lane];
      async_row_gather(x1 + (size_t)n * 64, &gbuf[w][(pp + 1) & 1][lane][0]);
      asm volatile("s_wait_asynccnt 0x8" ::: "memory");  // batch pp landed
    } else {
      asm volatile("s_wait_asynccnt 0x0" ::: "memory");
    }
    const int buf = pp & 1;
    const int pl = w * 4 + pp;
    const int g = gbase + pl;
    v16bf a = *(const v16bf*)(kpw + ((size_t)g * 32 + lane) * 16);
    #pragma unroll
    for (int ct = 0; ct < 4; ++ct) {
      v16bf bf;
      #pragma unroll
      for (int h = 0; h < 16; ++h) bf[h] = gbuf[w][buf][kb + h][ct * 16 + cl];
      v8f d = {};
      d = wmma_bf16(a, bf, d);
      #pragma unroll
      for (int r = 0; r < 8; ++r) {
        int Kp = r + 8 * hi;
        if (Kp < KPTS) agg[pl][Kp * 64 + ct * 16 + cl] = f2b(d[r]);
      }
    }
  }
  __syncthreads();

  // Stage 2: wave w owns output column tile w (pipelined global B loads)
  const int row = lane & 15;
  v8f acc = {};
  const __bf16* bb = wkp + ((size_t)w * 32 + lane) * 16;
  const size_t bstr = (size_t)4 * 512;  // ntiles=4
  v16bf bc = *(const v16bf*)(bb);
  for (int ks = 0; ks < 29; ++ks) {
    v16bf bn = *(const v16bf*)(bb + (size_t)(ks + 1) * bstr);
    int k0 = ks * 32;
    v8bf lo = *(const v8bf*)(&agg[row][k0 + hi * 8]);
    v8bf hs = *(const v8bf*)(&agg[row][k0 + 16 + hi * 8]);
    acc = wmma_bf16(cat16(lo, hs), bc, acc);
    bc = bn;
  }
  {
    v8bf lo = *(const v8bf*)(&agg[row][928 + hi * 8]);
    v8bf hs = *(const v8bf*)(&agg[row][944 + hi * 8]);
    acc = wmma_bf16(cat16(lo, hs), bc, acc);
  }
  #pragma unroll
  for (int r = 0; r < 8; ++r) {
    float v = acc[r];
    v = v > 0.f ? v : SLOPE * v;
    x2[(size_t)(gbase + r + 8 * hi) * 64 + w * 16 + cl] = f2b(v);
  }
}

// ---------------------------------------------------------------------------
// VLAD softmax assignment (wave per point), also writes transposed bf16 aT
// ---------------------------------------------------------------------------
__global__ __launch_bounds__(128)
void softmax_kernel(const float* __restrict__ logits,
                    const unsigned char* __restrict__ m,
                    float* __restrict__ a, __bf16* __restrict__ aT) {
  int lane = threadIdx.x & 31;
  int g = blockIdx.x * 4 + (threadIdx.x >> 5);
  int b = g >> 11, nl = g & (NN - 1);
  float v0 = logits[(size_t)g * 64 + lane];
  float v1 = logits[(size_t)g * 64 + 32 + lane];
  float mx = wredmax(fmaxf(v0, v1));
  float e0 = __expf(v0 - mx), e1 = __expf(v1 - mx);
  float s = wredsum(e0 + e1);
  float valid = (m[g] != 0) ? 0.f : 1.f;
  float a0 = e0 / s * valid, a1 = e1 / s * valid;
  a[(size_t)g * 64 + lane] = a0;
  a[(size_t)g * 64 + 32 + lane] = a1;
  aT[((size_t)b * 64 + lane) * NN + nl] = f2b(a0);
  aT[((size_t)b * 64 + 32 + lane) * NN + nl] = f2b(a1);
}

__global__ void zero_kernel(float* __restrict__ p, int n) {
  int i = blockIdx.x * blockDim.x + threadIdx.x;
  if (i < n) p[i] = 0.f;
}

// v = v0 - asum*centers ; intra-normalize over d(128); accumulate batch sumsq
__global__ __launch_bounds__(128)
void vlad_fix_kernel(const float* __restrict__ v0, const float* __restrict__ a,
                     const float* __restrict__ centers,
                     float* __restrict__ vn, float* __restrict__ bnorm) {
  int lane = threadIdx.x & 31;
  int rowid = blockIdx.x * 4 + (threadIdx.x >> 5);  // 0..511
  int b = rowid >> 6, k = rowid & 63;
  float s = 0.f;
  for (int n = lane; n < NN; n += 32) s += a[((size_t)b * NN + n) * 64 + k];
  s = wredsum(s);
  float vals[4]; float ss = 0.f;
  #pragma unroll
  for (int j = 0; j < 4; ++j) {
    int d = j * 32 + lane;
    float v = v0[((size_t)b * 64 + k) * 128 + d] - s * centers[(size_t)k * 128 + d];
    vals[j] = v; ss += v * v;
  }
  ss = wredsum(ss);
  float inv = 1.f / (sqrtf(ss) + 1e-8f);
  float bs = 0.f;
  #pragma unroll
  for (int j = 0; j < 4; ++j) {
    float wv = vals[j] * inv;
    vn[(size_t)b * 8192 + k * 128 + j * 32 + lane] = wv;
    bs += wv * wv;
  }
  bs = wredsum(bs);
  if (lane == 0) atomicAdd(&bnorm[b], bs);
}

// global flat-normalize, emit bf16 [16,8192] (rows 8..15 zero-padded for WMMA)
__global__ void vlad_norm_kernel(const float* __restrict__ vn,
                                 const float* __restrict__ bnorm,
                                 __bf16* __restrict__ vnbf) {
  int i = blockIdx.x * blockDim.x + threadIdx.x;
  if (i >= 16 * 8192) return;
  int b = i >> 13;
  float v = 0.f;
  if (b < BB) v = vn[(size_t)b * 8192 + (i & 8191)] / (sqrtf(bnorm[b]) + 1e-8f);
  vnbf[i] = f2b(v);
}

__global__ __launch_bounds__(256)
void final_norm_kernel(const float* __restrict__ pin, float* __restrict__ out) {
  __shared__ float red[256];
  int b = blockIdx.x, t = threadIdx.x;
  float v = pin[(size_t)b * 256 + t];
  red[t] = v * v;
  __syncthreads();
  for (int off = 128; off > 0; off >>= 1) {
    if (t < off) red[t] += red[t + off];
    __syncthreads();
  }
  out[(size_t)b * 256 + t] = v / (sqrtf(red[0]) + 1e-12f);
}

// ---------------------------------------------------------------------------
// Host driver
// ---------------------------------------------------------------------------
extern "C" void kernel_launch(void* const* d_in, const int* in_sizes, int n_in,
                              void* d_out, int out_size, void* d_ws, size_t ws_size,
                              hipStream_t stream) {
  (void)in_sizes; (void)n_in; (void)out_size; (void)ws_size;
  const float* x = (const float*)d_in[0];
  const unsigned char* m = (const unsigned char*)d_in[1];
  const float* pn_w1 = (const float*)d_in[2];
  const float* pn_b1 = (const float*)d_in[3];
  const float* pn_w2 = (const float*)d_in[4];
  const float* pn_b2 = (const float*)d_in[5];
  const float* kp = (const float*)d_in[6];
  const float* bw1[3] = {(const float*)d_in[7], (const float*)d_in[11], (const float*)d_in[15]};
  const float* bwk[3] = {(const float*)d_in[8], (const float*)d_in[12], (const float*)d_in[16]};
  const float* bw2[3] = {(const float*)d_in[9], (const float*)d_in[13], (const float*)d_in[17]};
  const float* bws[3] = {(const float*)d_in[10], (const float*)d_in[14], (const float*)d_in[18]};
  const float* wa = (const float*)d_in[19];
  const float* centers = (const float*)d_in[20];
  const float* proj = (const float*)d_in[21];
  const int cin[3] = {64, 128, 128};

  char* ws = (char*)d_ws;
  size_t off = 0;
  auto alloc = [&](size_t bytes) -> char* {
    off = (off + 255) & ~(size_t)255;
    char* p = ws + off;
    off += bytes;
    return p;
  };
  float*  coords = (float*)alloc((size_t)BN * 3 * 4);
  int*    idxb   = (int*)alloc((size_t)BN * KNBR * 4);
  __bf16* kpw    = (__bf16*)alloc((size_t)BN * 512 * 2);
  __bf16* f0bf   = (__bf16*)alloc((size_t)BN * 64 * 2);
  __bf16* featA  = (__bf16*)alloc((size_t)BN * 128 * 2);
  __bf16* featB  = (__bf16*)alloc((size_t)BN * 128 * 2);
  __bf16* x1bf   = (__bf16*)alloc((size_t)BN * 64 * 2);
  __bf16* x2bf   = (__bf16*)alloc((size_t)BN * 64 * 2);
  float*  featF  = (float*)alloc((size_t)BN * 128 * 4);
  float*  logits = (float*)alloc((size_t)BN * 64 * 4);
  float*  abuf   = (float*)alloc((size_t)BN * 64 * 4);
  __bf16* aT     = (__bf16*)alloc((size_t)BB * 64 * NN * 2);
  __bf16* fpack  = (__bf16*)alloc((size_t)BB * NN * 128 * 2);
  float*  v0buf  = (float*)alloc((size_t)BB * 64 * 128 * 4);
  float*  vn     = (float*)alloc((size_t)BB * 8192 * 4);
  float*  bnorm  = (float*)alloc(64 * 4);
  __bf16* vnbf   = (__bf16*)alloc((size_t)16 * 8192 * 2);
  float*  projO  = (float*)alloc((size_t)16 * 256 * 4);
  __bf16* w1p[3], *wkp[3], *w2p[3], *wsp[3];
  for (int l = 0; l < 3; ++l) {
    w1p[l] = (__bf16*)alloc((size_t)cin[l] * 64 * 2);
    wkp[l] = (__bf16*)alloc((size_t)960 * 64 * 2);
    w2p[l] = (__bf16*)alloc((size_t)64 * 128 * 2);
    wsp[l] = (__bf16*)alloc((size_t)cin[l] * 128 * 2);
  }
  __bf16* wap   = (__bf16*)alloc((size_t)128 * 64 * 2);
  __bf16* projp = (__bf16*)alloc((size_t)8192 * 256 * 2);

  auto pack = [&](const float* W, __bf16* o, int K, int N) {
    int total = (K / 32) * (N / 16) * 512;
    pack_b_kernel<<<(total + 255) / 256, 256, 0, stream>>>(W, o, K, N);
  };
  auto gemm = [&](const __bf16* A1, const __bf16* B1, int K1,
                  const __bf16* A2, const __bf16* B2, int K2,
                  int M, int N, int act, __bf16* Cbf, float* Cf) {
    int tiles = (M / 16) * (N / 16);
    gemm_bf16_kernel<<<(tiles + 3) / 4, 128, 0, stream>>>(A1, B1, K1, A2, B2, K2,
                                                          M, N, act, Cbf, Cf);
  };

  // --- weight packing (rerun each call; cheap, L2-resident) ---
  for (int l = 0; l < 3; ++l) {
    pack(bw1[l], w1p[l], cin[l], 64);
    pack(bwk[l], wkp[l], 960, 64);     // [15,64,64] flat == [960,64]
    pack(bw2[l], w2p[l], 64, 128);
    pack(bws[l], wsp[l], cin[l], 128);
  }
  pack(wa, wap, 128, 64);
  pack(proj, projp, 8192, 256);

  // --- geometry + pointnet ---
  coords_kernel<<<BN / 256, 256, 0, stream>>>(x, m, coords);
  pointnet_kernel<<<BN / 256, 256, 0, stream>>>(x, pn_w1, pn_b1, pn_w2, pn_b2, f0bf);
  knn_kernel<<<BN, 256, 0, stream>>>(coords, idxb);
  kpw_kernel<<<BN / 4, 128, 0, stream>>>(coords, idxb, kp, kpw);

  // --- 3 ResnetKPConv blocks ---
  const __bf16* fin = f0bf;
  __bf16* fout = featA;
  __bf16* fother = featB;
  for (int l = 0; l < 3; ++l) {
    gemm(fin, w1p[l], cin[l], nullptr, nullptr, 0, BN, 64, 1, x1bf, nullptr);
    kpconv_kernel<<<BN / 16, 128, 0, stream>>>(x1bf, idxb, kpw, wkp[l], x2bf);
    gemm(x2bf, w2p[l], 64, fin, wsp[l], cin[l], BN, 128, 1, fout, featF);
    fin = fout;
    __bf16* t = fout; fout = fother; fother = t;
  }

  // --- NetVLAD ---
  gemm(fin, wap, 128, nullptr, nullptr, 0, BN, 64, 0, nullptr, logits);
  softmax_kernel<<<BN / 4, 128, 0, stream>>>(logits, m, abuf, aT);
  for (int b = 0; b < BB; ++b) {
    pack(featF + (size_t)b * NN * 128, fpack + (size_t)b * NN * 128, NN, 128);
    gemm(aT + (size_t)b * 64 * NN, fpack + (size_t)b * NN * 128, NN,
         nullptr, nullptr, 0, 64, 128, 0, nullptr, v0buf + (size_t)b * 64 * 128);
  }
  zero_kernel<<<1, 64, 0, stream>>>(bnorm, BB);
  vlad_fix_kernel<<<(BB * 64) / 4, 128, 0, stream>>>(v0buf, abuf, centers, vn, bnorm);
  vlad_norm_kernel<<<(16 * 8192) / 256, 256, 0, stream>>>(vn, bnorm, vnbf);
  gemm(vnbf, projp, 8192, nullptr, nullptr, 0, 16, 256, 0, nullptr, projO);
  final_norm_kernel<<<BB, 256, 0, stream>>>(projO, (float*)d_out);
}
